// MambaBlock_28381143892421
// MI455X (gfx1250) — matrix operand order repaired and checked by reading, not compiled, not run on hardware
//
#include <hip/hip_runtime.h>

#define Bb 8
#define Ss 2048
#define Dd 512
#define Nn16 16
#define INNER 768
#define ROWS (Bb*Ss)

typedef __bf16 bf16_t;
typedef __attribute__((ext_vector_type(16))) __bf16 v16bf;
typedef __attribute__((ext_vector_type(8)))  __bf16 v8bf;
typedef __attribute__((ext_vector_type(8)))  float  v8f;

__device__ __forceinline__ unsigned short f2bf(float f) {
    unsigned int u = __float_as_uint(f);
    unsigned int r = (u + 0x7FFFu + ((u >> 16) & 1u)) >> 16;
    return (unsigned short)r;
}

// ---------------- weight conversion f32 -> bf16 (optionally padded stride) ---
__global__ void k_convert(const float* __restrict__ src, unsigned short* __restrict__ dst,
                          int rows, int cols, int dstStride) {
    int i = blockIdx.x * blockDim.x + threadIdx.x;
    if (i >= rows * cols) return;
    int r = i / cols, c = i % cols;
    dst[(size_t)r * dstStride + c] = f2bf(src[i]);
}

// ---------------- LayerNorm over D=512, one block per row ------------------
__global__ __launch_bounds__(256) void k_layernorm(
    const float* __restrict__ x, const float* __restrict__ g,
    const float* __restrict__ b, unsigned short* __restrict__ h) {
    int row = blockIdx.x, tid = threadIdx.x;
    const float* xr = x + (size_t)row * Dd;
    float x0 = xr[tid], x1 = xr[tid + 256];
    __shared__ float ss[256], sq[256];
    ss[tid] = x0 + x1; sq[tid] = x0 * x0 + x1 * x1;
    __syncthreads();
    for (int o = 128; o > 0; o >>= 1) {
        if (tid < o) { ss[tid] += ss[tid + o]; sq[tid] += sq[tid + o]; }
        __syncthreads();
    }
    float mu  = ss[0] * (1.f / Dd);
    float var = sq[0] * (1.f / Dd) - mu * mu;
    float inv = rsqrtf(var + 1e-5f);
    unsigned short* hr = h + (size_t)row * Dd;
    hr[tid]       = f2bf((x0 - mu) * inv * g[tid]       + b[tid]);
    hr[tid + 256] = f2bf((x1 - mu) * inv * g[tid + 256] + b[tid + 256]);
}

// ---------------- generic WMMA bf16 GEMM:  C = A(MxK) * W(NxK)^T -----------
// block = 8 waves; block tile 128(M) x (JT*16)(N); wave tile 16 x JT*16
// JT is compile-time -> inner K loop is pure loads + v_wmma, no control flow.
// MODE 0: proj -> data/gate stores + softplus-mean atomic into delta_acc
// MODE 1: wvec  -> f32 store (N=16, JT=1)
// MODE 2: selective -> gated_out = sigmoid(gate_in) * val   (bf16)
// MODE 3: out  -> out_final = xres + val
template <int MODE, int JT>
__global__ __launch_bounds__(256) void k_gemm(
    const unsigned short* __restrict__ A, const unsigned short* __restrict__ W,
    int Mdim, int Ndim, int Kdim,
    float* __restrict__ out_data, float* __restrict__ out_gate,
    float* __restrict__ delta_acc, float* __restrict__ out_f32,
    const float* __restrict__ gate_in, unsigned short* __restrict__ gated_out,
    const float* __restrict__ xres, float* __restrict__ out_final) {
    const int lane = threadIdx.x & 31;
    const int wid  = threadIdx.x >> 5;
    const int am   = blockIdx.x * 128 + wid * 16;
    const int bn0  = blockIdx.y * (JT * 16);
    const int mrow = am + (lane & 15);
    const int half = lane >> 4;

    v8f acc[JT];
#pragma unroll
    for (int j = 0; j < JT; j++)
#pragma unroll
        for (int e = 0; e < 8; e++) acc[j][e] = 0.f;

    const unsigned short* aRow = A + (size_t)mrow * Kdim + half * 16;
    const unsigned short* wPtr[JT];
#pragma unroll
    for (int j = 0; j < JT; j++)
        wPtr[j] = W + (size_t)(bn0 + j * 16 + (lane & 15)) * Kdim + half * 16;

    for (int k0 = 0; k0 < Kdim; k0 += 32) {
        v8bf lo = *(const v8bf*)(aRow + k0);
        v8bf hi = *(const v8bf*)(aRow + k0 + 8);
        v16bf afr = __builtin_shufflevector(lo, hi, 0, 1, 2, 3, 4, 5, 6, 7,
                                            8, 9, 10, 11, 12, 13, 14, 15);
#pragma unroll
        for (int j = 0; j < JT; j++) {
            v16bf bfr = *(const v16bf*)(wPtr[j] + k0);
            acc[j] = __builtin_amdgcn_wmma_f32_16x16x32_bf16(
                false, afr, false, bfr, (short)0, acc[j], false, false);
        }
    }

    if (MODE == 0) {
        int seg = bn0 / INNER;  // 0=data 1=gate 2=delta (64 | 768 so no straddle)
        if (seg < 2) {
            float* dst = (seg == 0) ? out_data : out_gate;
            int colbase = bn0 - seg * INNER;
#pragma unroll
            for (int j = 0; j < JT; j++) {
                int col = colbase + j * 16 + (lane & 15);
#pragma unroll
                for (int v = 0; v < 8; v++) {
                    int m = am + v + half * 8;
                    dst[(size_t)m * INNER + col] = acc[j][v];
                }
            }
        } else {
            float sp[8];
#pragma unroll
            for (int v = 0; v < 8; v++) sp[v] = 0.f;
#pragma unroll
            for (int j = 0; j < JT; j++)
#pragma unroll
                for (int v = 0; v < 8; v++) {
                    float xv = acc[j][v];
                    sp[v] += (xv > 20.f) ? xv : log1pf(__expf(xv));
                }
#pragma unroll
            for (int v = 0; v < 8; v++) {
                float p = sp[v] * (1.0f / INNER);
                p += __shfl_xor(p, 1, 32);
                p += __shfl_xor(p, 2, 32);
                p += __shfl_xor(p, 4, 32);
                p += __shfl_xor(p, 8, 32);
                if ((lane & 15) == 0)
                    atomicAdd(&delta_acc[am + v + half * 8], p);
            }
        }
    } else if (MODE == 1) {
#pragma unroll
        for (int j = 0; j < JT; j++) {
            int col = bn0 + j * 16 + (lane & 15);
#pragma unroll
            for (int v = 0; v < 8; v++) {
                int m = am + v + half * 8;
                out_f32[(size_t)m * Ndim + col] = acc[j][v];
            }
        }
    } else if (MODE == 2) {
#pragma unroll
        for (int j = 0; j < JT; j++) {
            int col = bn0 + j * 16 + (lane & 15);
#pragma unroll
            for (int v = 0; v < 8; v++) {
                int m = am + v + half * 8;
                float gv = gate_in[(size_t)m * INNER + col];
                float s  = 1.f / (1.f + __expf(-gv));
                gated_out[(size_t)m * INNER + col] = f2bf(acc[j][v] * s);
            }
        }
    } else {  // MODE 3
#pragma unroll
        for (int j = 0; j < JT; j++) {
            int col = bn0 + j * 16 + (lane & 15);
#pragma unroll
            for (int v = 0; v < 8; v++) {
                int m = am + v + half * 8;
                out_final[(size_t)m * Dd + col] = xres[(size_t)m * Dd + col] + acc[j][v];
            }
        }
    }
}

// ---------------- depthwise causal conv (K=3) + SiLU -> u (bf16) -----------
__global__ void k_conv(const float* __restrict__ data, const float* __restrict__ cw,
                       unsigned short* __restrict__ u) {
    size_t i = blockIdx.x * (size_t)blockDim.x + threadIdx.x;
    if (i >= (size_t)ROWS * INNER) return;
    int c = (int)(i % INNER);
    size_t row = i / INNER;
    int s = (int)(row % Ss);
    float acc = cw[c * 3 + 2] * data[row * INNER + c];
    if (s >= 1) acc += cw[c * 3 + 1] * data[(row - 1) * INNER + c];
    if (s >= 2) acc += cw[c * 3 + 0] * data[(row - 2) * INNER + c];
    float sig = 1.f / (1.f + __expf(-acc));
    u[row * INNER + c] = f2bf(acc * sig);
}

// ---------------- A^{-1} by Gauss-Jordan (A lower-tri, 16x16) --------------
__global__ __launch_bounds__(256) void k_ainv(const float* __restrict__ A,
                                              float* __restrict__ Ainv) {
    __shared__ float G[16][32];
    int tid = threadIdx.x;
    for (int e = tid; e < 512; e += 256) {
        int i = e >> 5, j = e & 31;
        G[i][j] = (j < 16) ? A[i * 16 + j] : (float)((j - 16) == i);
    }
    __syncthreads();
    for (int k = 0; k < 16; k++) {
        float piv = G[k][k];
        __syncthreads();
        for (int e = tid; e < 512; e += 256) {
            int i = e >> 5, j = e & 31;
            if (i == k) G[i][j] = G[i][j] / piv;
        }
        __syncthreads();
        float f[2]; int idx = 0;
        for (int e = tid; e < 512; e += 256) { f[idx++] = G[e >> 5][k]; }
        __syncthreads();
        idx = 0;
        for (int e = tid; e < 512; e += 256) {
            int i = e >> 5, j = e & 31;
            if (i != k) G[i][j] -= f[idx] * G[k][j];
            idx++;
        }
        __syncthreads();
    }
    for (int e = tid; e < 512; e += 256) {
        int i = e >> 5, j = e & 31;
        if (j >= 16) Ainv[i * 16 + (j - 16)] = G[i][j];
    }
}

// ---------------- expm(delta*A) via scale(2^-12)/Taylor(6)/square(12) ------
// also integral = Ainv*(E-I), v = integral * wvec
__global__ __launch_bounds__(256) void k_expm(
    const float* __restrict__ A, const float* __restrict__ Ainv,
    const float* __restrict__ delta_acc, const float* __restrict__ wvec,
    float* __restrict__ Ad, float* __restrict__ vbuf) {
    __shared__ float Msh[256], T[256], T2[256], Ash[256], wls[16];
    int tid = threadIdx.x;
    int i = tid >> 4, j = tid & 15;
    size_t row = blockIdx.x;
    float delta = fminf(delta_acc[row] + 1e-4f, 3.0f);
    float id = (i == j) ? 1.f : 0.f;
    Msh[tid] = A[tid] * (delta * (1.0f / 4096.0f));
    Ash[tid] = Ainv[tid];
    if (tid < 16) wls[tid] = wvec[row * 16 + tid];
    __syncthreads();
    T[tid] = id + Msh[tid] * (1.f / 6.f);
    __syncthreads();
#pragma unroll
    for (int k = 5; k >= 1; k--) {
        float s = 0.f;
#pragma unroll
        for (int kk = 0; kk < 16; kk++) s += Msh[i * 16 + kk] * T[kk * 16 + j];
        __syncthreads();
        T[tid] = id + s * (1.f / (float)k);
        __syncthreads();
    }
#pragma unroll
    for (int q = 0; q < 12; q++) {
        float s = 0.f;
#pragma unroll
        for (int kk = 0; kk < 16; kk++) s += T[i * 16 + kk] * T[kk * 16 + j];
        __syncthreads();
        T[tid] = s;
        __syncthreads();
    }
    Ad[row * 256 + tid] = T[tid];
    T[tid] -= id;
    __syncthreads();
    float s = 0.f;
#pragma unroll
    for (int kk = 0; kk < 16; kk++) s += Ash[i * 16 + kk] * T[kk * 16 + j];
    T2[tid] = s;
    __syncthreads();
    if (tid < 16) {
        float vv = 0.f;
#pragma unroll
        for (int m = 0; m < 16; m++) vv += T2[tid * 16 + m] * wls[m];
        vbuf[row * 16 + tid] = vv;
    }
}

// ---------------- sequential scan: one wave per batch ----------------------
// state_t = Ad_t @ state_{t-1} + v_t ; stores state (bf16, K padded to 32)
__global__ void k_scan(const float* __restrict__ Ad, const float* __restrict__ vbuf,
                       unsigned short* __restrict__ states) {
    __shared__ float st[16];
    int lane = threadIdx.x;
    int n = lane & 15, h = lane >> 4;
    if (lane < 16) st[lane] = 0.f;
    __syncthreads();
    size_t base = (size_t)blockIdx.x * Ss;
    for (int t = 0; t < Ss; t++) {
        size_t row = base + t;
        const float* ad = Ad + row * 256;
        if (t + 1 < Ss) __builtin_prefetch(Ad + (row + 1) * 256 + lane * 8, 0, 0);
        float4 a0 = *(const float4*)(ad + n * 16 + h * 8);
        float4 a1 = *(const float4*)(ad + n * 16 + h * 8 + 4);
        float p = a0.x * st[h * 8 + 0] + a0.y * st[h * 8 + 1] +
                  a0.z * st[h * 8 + 2] + a0.w * st[h * 8 + 3] +
                  a1.x * st[h * 8 + 4] + a1.y * st[h * 8 + 5] +
                  a1.z * st[h * 8 + 6] + a1.w * st[h * 8 + 7];
        p += __shfl_xor(p, 16, 32);
        __syncthreads();
        if (h == 0) {
            float ns = p + vbuf[row * 16 + n];
            st[n] = ns;
            states[row * 32 + n] = f2bf(ns);
        }
        __syncthreads();
    }
}

extern "C" void kernel_launch(void* const* d_in, const int* in_sizes, int n_in,
                              void* d_out, int out_size, void* d_ws, size_t ws_size,
                              hipStream_t stream) {
    const float* x      = (const float*)d_in[0];
    const float* ln_g   = (const float*)d_in[1];
    const float* ln_b   = (const float*)d_in[2];
    const float* W_in   = (const float*)d_in[3];
    const float* conv_w = (const float*)d_in[4];
    const float* Aptr   = (const float*)d_in[5];
    const float* Bmat   = (const float*)d_in[6];
    const float* Cmat   = (const float*)d_in[7];
    const float* W_out  = (const float*)d_in[8];
    float* out = (float*)d_out;

    char* ws = (char*)d_ws;
    size_t off = 0;
    auto alloc = [&](size_t bytes) -> void* {
        void* p = ws + off;
        off = (off + bytes + 255) & ~(size_t)255;
        return p;
    };
    unsigned short* h_bf    = (unsigned short*)alloc((size_t)ROWS * Dd * 2);
    unsigned short* Win_bf  = (unsigned short*)alloc((size_t)3 * INNER * Dd * 2);
    unsigned short* Wout_bf = (unsigned short*)alloc((size_t)Dd * INNER * 2);
    unsigned short* Bm_bf   = (unsigned short*)alloc((size_t)Nn16 * INNER * 2);
    unsigned short* Cm_bf   = (unsigned short*)alloc((size_t)INNER * 32 * 2);
    float* data_f           = (float*)alloc((size_t)ROWS * INNER * 4);
    float* gate_f           = (float*)alloc((size_t)ROWS * INNER * 4);
    float* delta_a          = (float*)alloc((size_t)ROWS * 4);
    unsigned short* u_bf    = (unsigned short*)alloc((size_t)ROWS * INNER * 2);
    float* wvec_f           = (float*)alloc((size_t)ROWS * Nn16 * 4);
    float* Ad_f             = (float*)alloc((size_t)ROWS * 256 * 4);
    float* v_f              = (float*)alloc((size_t)ROWS * Nn16 * 4);
    float* Ainv_f           = (float*)alloc(256 * 4);
    unsigned short* st_bf   = (unsigned short*)alloc((size_t)ROWS * 32 * 2);
    unsigned short* gated_bf= (unsigned short*)alloc((size_t)ROWS * INNER * 2);

    hipMemsetAsync(delta_a, 0, (size_t)ROWS * 4, stream);
    hipMemsetAsync(st_bf, 0, (size_t)ROWS * 32 * 2, stream);
    hipMemsetAsync(Cm_bf, 0, (size_t)INNER * 32 * 2, stream);

    int n1 = 3 * INNER * Dd;
    k_convert<<<(n1 + 255) / 256, 256, 0, stream>>>(W_in, Win_bf, 3 * INNER, Dd, Dd);
    int n2 = Dd * INNER;
    k_convert<<<(n2 + 255) / 256, 256, 0, stream>>>(W_out, Wout_bf, Dd, INNER, INNER);
    int n3 = Nn16 * INNER;
    k_convert<<<(n3 + 255) / 256, 256, 0, stream>>>(Bmat, Bm_bf, Nn16, INNER, INNER);
    int n4 = INNER * Nn16;
    k_convert<<<(n4 + 255) / 256, 256, 0, stream>>>(Cmat, Cm_bf, INNER, Nn16, 32);

    k_ainv<<<1, 256, 0, stream>>>(Aptr, Ainv_f);
    k_layernorm<<<ROWS, 256, 0, stream>>>(x, ln_g, ln_b, h_bf);

    dim3 g0(ROWS / 128, (3 * INNER) / 64);
    k_gemm<0, 4><<<g0, 256, 0, stream>>>(h_bf, Win_bf, ROWS, 3 * INNER, Dd,
                                         data_f, gate_f, delta_a, nullptr,
                                         nullptr, nullptr, nullptr, nullptr);

    size_t nc = (size_t)ROWS * INNER;
    k_conv<<<(unsigned)((nc + 255) / 256), 256, 0, stream>>>(data_f, conv_w, u_bf);

    dim3 g1(ROWS / 128, 1);
    k_gemm<1, 1><<<g1, 256, 0, stream>>>(u_bf, Bm_bf, ROWS, Nn16, INNER,
                                         nullptr, nullptr, nullptr, wvec_f,
                                         nullptr, nullptr, nullptr, nullptr);

    k_expm<<<ROWS, 256, 0, stream>>>(Aptr, Ainv_f, delta_a, wvec_f, Ad_f, v_f);
    k_scan<<<Bb, 32, 0, stream>>>(Ad_f, v_f, st_bf);

    dim3 g2(ROWS / 128, INNER / 64);
    k_gemm<2, 4><<<g2, 256, 0, stream>>>(st_bf, Cm_bf, ROWS, INNER, 32,
                                         nullptr, nullptr, nullptr, nullptr,
                                         gate_f, gated_bf, nullptr, nullptr);

    dim3 g3(ROWS / 128, Dd / 64);
    k_gemm<3, 4><<<g3, 256, 0, stream>>>(gated_bf, Wout_bf, ROWS, Dd, INNER,
                                         nullptr, nullptr, nullptr, nullptr,
                                         nullptr, nullptr, x, out);
}
